// MixedResolutionCNN_89644557402566
// MI455X (gfx1250) — matrix operand order repaired and checked by reading, not compiled, not run on hardware
//
#include <hip/hip_runtime.h>
#include <hip/hip_bf16.h>

// ---------- types ----------
typedef __attribute__((ext_vector_type(8)))  unsigned short v8us;
typedef __attribute__((ext_vector_type(16))) unsigned short v16us;
typedef __attribute__((ext_vector_type(16))) __bf16        v16bf;
typedef __attribute__((ext_vector_type(8)))  float         v8f;
typedef __attribute__((ext_vector_type(4)))  int           v4i;
typedef __attribute__((address_space(1))) v4i v4i_g;   // global
typedef __attribute__((address_space(3))) v4i v4i_l;   // LDS

// ---------- helpers ----------
__device__ __forceinline__ unsigned short f2us(float f) {
    unsigned u = __builtin_bit_cast(unsigned, f);
    u += 0x7FFFu + ((u >> 16) & 1u);          // round-to-nearest-even bf16
    return (unsigned short)(u >> 16);
}

// A-matrix fragment (16x32 bf16): lane L holds row M=L%16.
// lanes 0-15: K = k0+0..15
// lanes 16-31: K = k0+16..31
__device__ __forceinline__ v16bf ldA(const unsigned short* base, int ld, int k0, int lane) {
    const unsigned short* p = base + (lane & 15) * ld + k0 + ((lane & 16) ? 16 : 0);
    v8us c1 = *(const v8us*)p;
    v8us c2 = *(const v8us*)(p + 8);
    v16us w = __builtin_shufflevector(c1, c2, 0,1,2,3,4,5,6,7,8,9,10,11,12,13,14,15);
    return __builtin_bit_cast(v16bf, w);
}

// B-matrix fragment (32x16 bf16), source stored as B^T row-major (i.e. weight rows).
// lane L holds column N=L%16; lanes 0-15: K=k0..k0+15, lanes 16-31: K=k0+16..k0+31.
__device__ __forceinline__ v16bf ldB(const unsigned short* base, int ld, int k0, int lane) {
    const unsigned short* p = base + (lane & 15) * ld + k0 + ((lane & 16) ? 16 : 0);
    v16us w = *(const v16us*)p;
    return __builtin_bit_cast(v16bf, w);
}

__device__ __forceinline__ v8f wmma_bf16(v16bf a, v16bf b, v8f c) {
    return __builtin_amdgcn_wmma_f32_16x16x32_bf16(
        /*neg_a=*/false, a, /*neg_b=*/false, b,
        /*c_mod=*/(short)0, c, /*reuse_a=*/false, /*reuse_b=*/false);
}

// CDNA5 async global->LDS copy (16 bytes per lane), guarded so the build can't break.
__device__ __forceinline__ void async_cp16(const void* g, void* l) {
#if __has_builtin(__builtin_amdgcn_global_load_async_to_lds_b128)
    __builtin_amdgcn_global_load_async_to_lds_b128((v4i_g*)g, (v4i_l*)l, 0, 0);
#else
    *(v8us*)l = *(const v8us*)g;
#endif
}
__device__ __forceinline__ void async_wait0() {
#if __has_builtin(__builtin_amdgcn_global_load_async_to_lds_b128) && \
    __has_builtin(__builtin_amdgcn_s_wait_asynccnt)
    __builtin_amdgcn_s_wait_asynccnt(0);
#endif
}

// ---------- sizes ----------
#define BATCH   4096
#define FCIN    4160          // 64 low + 4096 high
#define PADW    48            // 32 + 2*7 pad, +2 spare zero rows/cols
#define PIMG    (PADW*PADW)   // 2304

// ---------- prep: ridgelet kernel -> bf16 A matrix [16][256], tap t = ky*16+kx ----------
// rows ky==15 and cols kx==15 are exactly zero (padding), which the conv kernel exploits.
__global__ void k_ridgelet(const float* __restrict__ dirs, const float* __restrict__ scales,
                           const float* __restrict__ pos, unsigned short* __restrict__ kernA) {
    int oc = blockIdx.x, t = threadIdx.x;
    int ky = t >> 4, kx = t & 15;
    float v = 0.f;
    if (ky < 15 && kx < 15) {
        float x1 = (float)ky - 7.f, x2 = (float)kx - 7.f;
        for (int j = 0; j < 10; ++j) {
            int idx = oc * 10 + j;
            float d = dirs[idx], s = scales[idx], p = pos[idx];
            float tt = (x1 * __cosf(d) + x2 * __sinf(d) - p) / s;
            float q = tt * tt;
            v += __expf(-0.5f * q) - 0.5f * __expf(-0.125f * q);
        }
    }
    kernA[oc * 256 + t] = f2us(v);
}

// ---------- prep: channel-sum + zero-pad image -> bf16 [B][48][48] ----------
__global__ void k_xpad(const float* __restrict__ x, unsigned short* __restrict__ xpad) {
    int i = blockIdx.x * 256 + threadIdx.x;          // exact: 4096*2304
    int b = i / PIMG, rem = i % PIMG;
    int y = rem / PADW, c = rem % PADW;
    float v = 0.f;
    if (y >= 7 && y < 39 && c >= 7 && c < 39) {
        const float* p = x + (size_t)b * 3 * 1024 + (y - 7) * 32 + (c - 7);
        v = p[0] + p[1024] + p[2048];
    }
    xpad[i] = f2us(v);
}

// ---------- prep: f32 -> bf16 cast ----------
__global__ void k_cast(const float* __restrict__ s, unsigned short* __restrict__ d, int n) {
    int i = blockIdx.x * 256 + threadIdx.x;
    if (i < n) d[i] = f2us(s[i]);
}

// ---------- prep: w3 (5,256) -> padded bf16 (16,256) ----------
__global__ void k_w3pad(const float* __restrict__ w3, unsigned short* __restrict__ d) {
    int i = blockIdx.x * 256 + threadIdx.x;          // 16*256
    int oc = i >> 8, k = i & 255;
    d[i] = (oc < 5) ? f2us(w3[oc * 256 + k]) : (unsigned short)0;
}

// ---------- low path: bilinear /4 -> conv3x3+relu -> pool2 -> h[b][0..63] ----------
__global__ void __launch_bounds__(64) k_low(const float* __restrict__ x,
                                            const float* __restrict__ wlow,
                                            const float* __restrict__ blow,
                                            unsigned short* __restrict__ h) {
    __shared__ float xl[3][8][8];
    __shared__ float xc[4][8][8];
    int b = blockIdx.x, t = threadIdx.x;
    int oy = t >> 3, ox = t & 7;
    const float* xb = x + (size_t)b * 3 * 1024;
    // bilinear, scale 4, half-pixel: src = 4*i + 1.5 -> mean of pixels 4i+1,4i+2
    for (int c = 0; c < 3; ++c) {
        const float* p = xb + c * 1024 + (4 * oy + 1) * 32 + (4 * ox + 1);
        xl[c][oy][ox] = 0.25f * (p[0] + p[1] + p[32] + p[33]);
    }
    __syncthreads();
    for (int oc = 0; oc < 4; ++oc) {
        float s = blow[oc];
        for (int c = 0; c < 3; ++c)
            for (int ky = 0; ky < 3; ++ky) {
                int iy = oy + ky - 1;
                if (iy < 0 || iy > 7) continue;
                for (int kx = 0; kx < 3; ++kx) {
                    int ix = ox + kx - 1;
                    if (ix < 0 || ix > 7) continue;
                    s += xl[c][iy][ix] * wlow[((oc * 3 + c) * 3 + ky) * 3 + kx];
                }
            }
        xc[oc][oy][ox] = fmaxf(s, 0.f);
    }
    __syncthreads();
    int oc = t >> 4, py = (t >> 2) & 3, px = t & 3;
    float m = fmaxf(fmaxf(xc[oc][2*py][2*px],   xc[oc][2*py][2*px+1]),
                    fmaxf(xc[oc][2*py+1][2*px], xc[oc][2*py+1][2*px+1]));
    h[(size_t)b * FCIN + oc * 16 + py * 4 + px] = f2us(m);
}

// ---------- high path conv as implicit GEMM (WMMA), relu + pool fused ----------
// wave tile: 16 out-channels x 16 x-positions x 2 rows; K = 256 taps (15x16 padded + zero row).
// One shared 16x32 strip serves both output rows: zero-weight taps (ky==15, kx==15) may read
// arbitrary *finite* image data, so row clamping replaces zero-fill.
__global__ void __launch_bounds__(128) k_conv(const unsigned short* __restrict__ xpad,
                                              const unsigned short* __restrict__ kernA,
                                              unsigned short* __restrict__ h) {
    __shared__ __align__(32) unsigned short strip[4][16][32];
    int w = threadIdx.x >> 5, lane = threadIdx.x & 31;
    int gw = blockIdx.x * 4 + w;                     // exact: 4096*32 waves
    int b = gw >> 5, rem = gw & 31;
    int py = rem >> 1, x0 = (rem & 1) << 4;

    // stage 16x32 input strip (img rows 2py..2py+15) via async copy to LDS
    const unsigned short* img = xpad + (size_t)b * PIMG + (2 * py) * PADW + x0;
    {
        int r = lane >> 1, c = (lane & 1) << 4;      // 32 lanes cover 16 rows x 2 halves
        const unsigned short* src = img + r * PADW + c;
        unsigned short* dst = &strip[w][r][c];
        async_cp16(src, dst);
        async_cp16(src + 8, dst + 8);
    }

    // preload all 8 weight A-fragments (shared by every wave, L2-resident)
    v16bf a[8];
#pragma unroll
    for (int ks = 0; ks < 8; ++ks) a[ks] = ldA(kernA, 256, ks * 32, lane);

    async_wait0();
    __syncthreads();

    v8f acc0 = {}, acc1 = {};
    int n = lane & 15;
#pragma unroll
    for (int ks = 0; ks < 8; ++ks) {
        // each aligned 16-K run lies in one kernel row: ky = 2*ks + (lane>=16), kx = j
        int ky0 = (ks << 1) + ((lane & 16) ? 1 : 0);       // 0..15
        int ky1 = (ky0 + 1 < 16) ? ky0 + 1 : 15;          // clamp: weight row 15 is zero
        v16us b0, b1;
#pragma unroll
        for (int j = 0; j < 16; ++j) {
            b0[j] = strip[w][ky0][n + j];
            b1[j] = strip[w][ky1][n + j];
        }
        acc0 = wmma_bf16(a[ks], __builtin_bit_cast(v16bf, b0), acc0);
        acc1 = wmma_bf16(a[ks], __builtin_bit_cast(v16bf, b1), acc1);
    }

    // relu + 2x2 maxpool (vertical in-regs, horizontal via shfl_xor), store bf16 into concat h
#pragma unroll
    for (int r = 0; r < 8; ++r) {
        float v = fmaxf(fmaxf(acc0[r], acc1[r]), 0.0f);
        float o = fmaxf(v, __shfl_xor(v, 1, 32));
        if ((lane & 1) == 0) {
            int oc = r + ((lane & 16) ? 8 : 0);
            int px = (x0 >> 1) + (n >> 1);
            h[(size_t)b * FCIN + 64 + oc * 256 + py * 16 + px] = f2us(o);
        }
    }
}

// ---------- generic FC: out = relu(A @ B^T + bias) in bf16; wave tile 32x64 ----------
__global__ void __launch_bounds__(128) k_fc(const unsigned short* __restrict__ A,
                                            const unsigned short* __restrict__ B,
                                            const float* __restrict__ bias,
                                            unsigned short* __restrict__ out,
                                            int K, int N) {
    int lane = threadIdx.x & 31;
    int gw = (blockIdx.x * 128 + threadIdx.x) >> 5;
    int nt64 = N >> 6;
    int mt = gw / nt64, nt = gw % nt64;
    int mbase = mt << 5, nbase = nt << 6;
    const unsigned short* A0 = A + (size_t)mbase * K;
    const unsigned short* A1 = A0 + (size_t)16 * K;
    v8f acc[2][4] = {};
    for (int k0 = 0; k0 < K; k0 += 32) {
        __builtin_prefetch((const void*)(A0 + (lane & 15) * K + k0 + 256), 0, 0);
        v16bf a0 = ldA(A0, K, k0, lane);
        v16bf a1 = ldA(A1, K, k0, lane);
#pragma unroll
        for (int j = 0; j < 4; ++j) {
            v16bf bfrag = ldB(B + (size_t)(nbase + 16 * j) * K, K, k0, lane);
            acc[0][j] = wmma_bf16(a0, bfrag, acc[0][j]);
            acc[1][j] = wmma_bf16(a1, bfrag, acc[1][j]);
        }
    }
    int n = lane & 15, half = (lane & 16) ? 8 : 0;
#pragma unroll
    for (int mi = 0; mi < 2; ++mi)
#pragma unroll
        for (int j = 0; j < 4; ++j) {
            float bv = bias[nbase + 16 * j + n];
#pragma unroll
            for (int r = 0; r < 8; ++r) {
                float v = fmaxf(acc[mi][j][r] + bv, 0.0f);
                out[(size_t)(mbase + mi * 16 + r + half) * N + nbase + 16 * j + n] = f2us(v);
            }
        }
}

// ---------- final FC: (4096x256) @ (5x256)^T + b3 -> f32, via padded 16-col WMMA ----------
__global__ void __launch_bounds__(128) k_fc3(const unsigned short* __restrict__ A,
                                             const unsigned short* __restrict__ B,
                                             const float* __restrict__ b3,
                                             float* __restrict__ out) {
    int lane = threadIdx.x & 31;
    int gw = (blockIdx.x * 128 + threadIdx.x) >> 5;
    int mbase = gw << 4;
    v8f acc = {};
#pragma unroll
    for (int ks = 0; ks < 8; ++ks) {
        v16bf af = ldA(A + (size_t)mbase * 256, 256, ks * 32, lane);
        v16bf bfrag = ldB(B, 256, ks * 32, lane);
        acc = wmma_bf16(af, bfrag, acc);
    }
    int n = lane & 15, half = (lane & 16) ? 8 : 0;
    if (n < 5) {
        float bv = b3[n];
#pragma unroll
        for (int r = 0; r < 8; ++r)
            out[(size_t)(mbase + r + half) * 5 + n] = acc[r] + bv;
    }
}

// ---------- launcher ----------
extern "C" void kernel_launch(void* const* d_in, const int* in_sizes, int n_in,
                              void* d_out, int out_size, void* d_ws, size_t ws_size,
                              hipStream_t stream) {
    const float* x        = (const float*)d_in[0];
    const float* r_scales = (const float*)d_in[1];
    const float* r_dirs   = (const float*)d_in[2];
    const float* r_pos    = (const float*)d_in[3];
    const float* wlow     = (const float*)d_in[4];
    const float* blow     = (const float*)d_in[5];
    const float* w1       = (const float*)d_in[6];
    const float* b1       = (const float*)d_in[7];
    const float* w2       = (const float*)d_in[8];
    const float* b2       = (const float*)d_in[9];
    const float* w3       = (const float*)d_in[10];
    const float* b3       = (const float*)d_in[11];
    float* out = (float*)d_out;

    char* ws = (char*)d_ws;
    unsigned short* kernA = (unsigned short*)(ws + 0);            //   8 KB
    unsigned short* w3pad = (unsigned short*)(ws + 8192);         //   8 KB
    unsigned short* xpad  = (unsigned short*)(ws + 16384);        //  18.9 MB (4096*2304)
    unsigned short* h     = (unsigned short*)(ws + 18890752ull);  //  34.1 MB (4096*4160)
    unsigned short* w1b   = (unsigned short*)(ws + 52969472ull);  //   8.5 MB (1024*4160)
    unsigned short* h1    = (unsigned short*)(ws + 61489152ull);  //   8.4 MB (4096*1024)
    unsigned short* w2b   = (unsigned short*)(ws + 69877760ull);  //   0.5 MB (256*1024)
    unsigned short* h2    = (unsigned short*)(ws + 70402048ull);  //   2.1 MB (4096*256)

    // prep
    k_ridgelet<<<16, 256, 0, stream>>>(r_dirs, r_scales, r_pos, kernA);
    k_w3pad   <<<16, 256, 0, stream>>>(w3, w3pad);
    k_cast    <<<(1024 * 4160 + 255) / 256, 256, 0, stream>>>(w1, w1b, 1024 * 4160);
    k_cast    <<<(256 * 1024 + 255) / 256, 256, 0, stream>>>(w2, w2b, 256 * 1024);
    k_xpad    <<<(BATCH * PIMG) / 256, 256, 0, stream>>>(x, xpad);

    // low path -> h[:, 0:64]
    k_low<<<BATCH, 64, 0, stream>>>(x, wlow, blow, h);

    // high path conv (WMMA implicit GEMM) -> h[:, 64:4160]
    k_conv<<<(BATCH * 32) / 4, 128, 0, stream>>>(xpad, kernA, h);

    // FC stack (32x64 wave tiles)
    k_fc <<<(128 * 16) / 4, 128, 0, stream>>>(h,  w1b, b1, h1, FCIN, 1024);  // 4096x4160 -> 4096x1024
    k_fc <<<(128 * 4)  / 4, 128, 0, stream>>>(h1, w2b, b2, h2, 1024, 256);   // -> 4096x256
    k_fc3<<<256 / 4, 128, 0, stream>>>(h2, w3pad, b3, out);                  // -> 4096x5 f32
}
